// BPPSModel_43791486550212
// MI455X (gfx1250) — matrix operand order repaired and checked by reading, not compiled, not run on hardware
//
#include <hip/hip_runtime.h>
#include <math.h>

#define N_ATOMS   20000
#define N_EDGES   640000
#define N_STRUCT  100
#define N_SPECIES 4
#define N_RADIAL  8
#define NSPH      16
#define QDIM      32          // N_SPECIES * N_RADIAL
#define PS_DIM    4096        // 4 * 32 * 32
#define HID       256
#define CUTOFF    5.0f
#define EPSF      1e-12f
#define PI_F      3.14159265358979323846f
// species groups padded to 32 rows so every 32-row wave tile is species-uniform;
// worst case 20000 + 4*31 = 20124 -> round order array to 315*64 = 20160
#define ORDER_PAD  20160
#define GEMM_GRID  (ORDER_PAD / 64)            // 315 blocks, 4 mtiles (64 rows) each

typedef __attribute__((ext_vector_type(16))) __bf16 v16bf;
typedef __attribute__((ext_vector_type(8)))  __bf16 v8bf;
typedef __attribute__((ext_vector_type(8)))  float  v8f;

__device__ __forceinline__ unsigned short f32_to_bf16(float f) {
  unsigned int u = __float_as_uint(f);
  u += 0x7FFFu + ((u >> 16) & 1u);          // round-to-nearest-even
  return (unsigned short)(u >> 16);
}
__device__ __forceinline__ float bf16_to_f32(unsigned short h) {
  return __uint_as_float(((unsigned int)h) << 16);
}

// ---------------------------------------------------------------- utilities
__global__ void zero_f32_kernel(float* __restrict__ p, int n) {
  int i = blockIdx.x * 256 + threadIdx.x;
  if (i < n) p[i] = 0.0f;
}

__global__ void init_meta_order_kernel(int* __restrict__ meta, int* __restrict__ order) {
  int i = blockIdx.x * 256 + threadIdx.x;
  if (i < ORDER_PAD) order[i] = -1;
  if (i < 16) meta[i] = 0;
}

__global__ void bucket_count_kernel(const int* __restrict__ numbers, int* __restrict__ meta) {
  int a = blockIdx.x * 256 + threadIdx.x;
  if (a < N_ATOMS) atomicAdd(&meta[numbers[a]], 1);
}

__global__ void bucket_offsets_kernel(int* __restrict__ meta) {
  if (blockIdx.x == 0 && threadIdx.x == 0) {
    int off = 0;
    for (int s = 0; s < N_SPECIES; ++s) {
      meta[4 + s] = off;
      off += ((meta[s] + 31) / 32) * 32;   // pad each species group to a 32-row pair-tile
      meta[9 + s] = 0;                      // cursors
    }
    meta[8] = off;                          // padded total
  }
}

__global__ void bucket_fill_kernel(const int* __restrict__ numbers,
                                   int* __restrict__ meta, int* __restrict__ order) {
  int a = blockIdx.x * 256 + threadIdx.x;
  if (a < N_ATOMS) {
    int s = numbers[a];
    int pos = meta[4 + s] + atomicAdd(&meta[9 + s], 1);
    order[pos] = a;
  }
}

// ---------------------------------------------------------------- edges -> c
__global__ void edge_kernel(const float* __restrict__ pos, const float* __restrict__ cells,
                            const int* __restrict__ numbers, const int* __restrict__ ei,
                            const int* __restrict__ eoff, const int* __restrict__ batch,
                            float* __restrict__ c) {
  int e = blockIdx.x * 256 + threadIdx.x;
  if (e >= N_EDGES) return;
  int i = ei[e];
  int j = ei[N_EDGES + e];
  const float* cell = cells + batch[i] * 9;
  float o0 = (float)eoff[e * 3 + 0], o1 = (float)eoff[e * 3 + 1], o2 = (float)eoff[e * 3 + 2];
  float sx = o0 * cell[0] + o1 * cell[3] + o2 * cell[6];
  float sy = o0 * cell[1] + o1 * cell[4] + o2 * cell[7];
  float sz = o0 * cell[2] + o1 * cell[5] + o2 * cell[8];
  float rx = pos[j * 3 + 0] - pos[i * 3 + 0] + sx;
  float ry = pos[j * 3 + 1] - pos[i * 3 + 1] + sy;
  float rz = pos[j * 3 + 2] - pos[i * 3 + 2] + sz;
  float r = sqrtf(rx * rx + ry * ry + rz * rz + EPSF);
  float fc = (r < CUTOFF) ? 0.5f * (__cosf(PI_F * r / CUTOFF) + 1.0f) : 0.0f;
  float inv = 1.0f / r;
  float x = rx * inv, y = ry * inv, z = rz * inv;
  float x2 = x * x, y2 = y * y, z2 = z * z;

  float Y[16];
  Y[0]  = 0.28209479177387814f;
  Y[1]  = 0.4886025119029199f * y;
  Y[2]  = 0.4886025119029199f * z;
  Y[3]  = 0.4886025119029199f * x;
  Y[4]  = 1.0925484305920792f * x * y;
  Y[5]  = 1.0925484305920792f * y * z;
  Y[6]  = 0.31539156525252005f * (3.0f * z2 - 1.0f);
  Y[7]  = 1.0925484305920792f * x * z;
  Y[8]  = 0.5462742152960396f * (x2 - y2);
  Y[9]  = 0.5900435899266435f * y * (3.0f * x2 - y2);
  Y[10] = 2.890611442640554f  * x * y * z;
  Y[11] = 0.4570457994644658f * y * (5.0f * z2 - 1.0f);
  Y[12] = 0.3731763325901154f * z * (5.0f * z2 - 3.0f);
  Y[13] = 0.4570457994644658f * x * (5.0f * z2 - 1.0f);
  Y[14] = 1.445305721320277f  * z * (x2 - y2);
  Y[15] = 0.5900435899266435f * x * (x2 - 3.0f * y2);

  float arg = PI_F * r / CUTOFF;
  int q0 = numbers[j] * N_RADIAL;
  float* cb = c + (size_t)i * (QDIM * NSPH);
  #pragma unroll
  for (int n = 0; n < N_RADIAL; ++n) {
    float rn = __sinf((float)(n + 1) * arg) * inv * fc;
    float* dst = cb + (q0 + n) * NSPH;
    #pragma unroll
    for (int m = 0; m < NSPH; ++m) unsafeAtomicAdd(dst + m, rn * Y[m]);
  }
}

// ---------------------------------------------------------------- power spectrum -> bf16
__global__ void ps_kernel(const float* __restrict__ c, unsigned short* __restrict__ psB) {
  const float scl[4] = {1.0f, 0.5773502691896258f, 0.4472135954999579f, 0.3779644730092272f};
  int atom = blockIdx.x;
  __shared__ float cs[QDIM * NSPH];
  for (int t = threadIdx.x; t < QDIM * NSPH; t += 256)
    cs[t] = c[(size_t)atom * (QDIM * NSPH) + t];
  __syncthreads();
  #pragma unroll
  for (int it = 0; it < PS_DIM / 256; ++it) {
    int idx = it * 256 + threadIdx.x;
    int l = idx >> 10;
    int q = (idx >> 5) & 31;
    int p = idx & 31;
    int m0 = l * l, m1 = (l + 1) * (l + 1);
    float sum = 0.0f;
    for (int m = m0; m < m1; ++m) sum += cs[q * NSPH + m] * cs[p * NSPH + m];
    psB[(size_t)atom * PS_DIM + idx] = f32_to_bf16(sum * scl[l]);
  }
}

// ---------------------------------------------------------------- weights: [s][K][N] f32 -> [s][N][K] bf16
__global__ void w_transpose_bf16_kernel(const float* __restrict__ W,
                                        unsigned short* __restrict__ Wt, int K, int N) {
  long idx = (long)blockIdx.x * 256 + threadIdx.x;
  long total = (long)N_SPECIES * K * N;
  if (idx >= total) return;
  int n = (int)(idx % N);
  long t = idx / N;
  int k = (int)(t % K);
  int s = (int)(t / K);
  Wt[((long)s * N + n) * K + k] = f32_to_bf16(W[idx]);
}

// ---------------------------------------------------------------- WMMA GEMM (+SiLU, bf16 out)
// Block = 256 threads = 8 waves. Each wave: a 32-row x 64-col tile (2 M-tiles x 4 N-tiles,
// 8 f32 accumulators). Per 32-deep K step: 4 A loads + 8 B loads feed 8 WMMAs (1.5 vmem/wmma),
// with B fragments reused across both M-tiles in registers. Species groups are padded to 32
// rows so the whole wave tile shares one weight matrix.
__global__ void gemm_species_silu_kernel(const unsigned short* __restrict__ Aus,
                                         const unsigned short* __restrict__ Wtus,
                                         unsigned short* __restrict__ Out,
                                         const int* __restrict__ order,
                                         const int* __restrict__ meta, int K) {
  const int lane  = threadIdx.x & 31;
  const int wave  = threadIdx.x >> 5;
  const int mpair = blockIdx.x * 2 + (wave >> 2);
  const int nbase = (wave & 3) * 64;
  const int row0  = mpair * 32;

  int s = 0;
  if (row0 >= meta[5]) s = 1;
  if (row0 >= meta[6]) s = 2;
  if (row0 >= meta[7]) s = 3;

  const __bf16* A = (const __bf16*)Aus;
  const __bf16* W = (const __bf16*)Wtus + (size_t)s * HID * K;

  const int half = lane >> 4;           // lanes 0-15 vs 16-31
  const int mA = lane & 15;
  int arow0 = order[row0 + mA];       if (arow0 < 0) arow0 = 0;  // padded rows: load row 0,
  int arow1 = order[row0 + 16 + mA];  if (arow1 < 0) arow1 = 0;  // store skipped below
  const __bf16* Arow0 = A + (size_t)arow0 * K + half * 16;
  const __bf16* Arow1 = A + (size_t)arow1 * K + half * 16;

  const int nB = lane & 15;
  const __bf16* Wl[4];
  #pragma unroll
  for (int nt = 0; nt < 4; ++nt)
    Wl[nt] = W + (size_t)(nbase + nt * 16 + nB) * K + half * 16;

  v8f acc[8];
  #pragma unroll
  for (int t = 0; t < 8; ++t)
    #pragma unroll
    for (int r = 0; r < 8; ++r) acc[t][r] = 0.0f;

  for (int k0 = 0; k0 < K; k0 += 32) {
    // A is the only HBM-streamed operand (weights sit in L2): prefetch ~2KB ahead.
    __builtin_prefetch(Arow0 + k0 + 1024, 0, 1);
    __builtin_prefetch(Arow1 + k0 + 1024, 0, 1);

    v8bf a00 = *(const v8bf*)(Arow0 + k0);        // K = half*16 + 0..7
    v8bf a01 = *(const v8bf*)(Arow0 + k0 + 8);    // K = half*16 + 8..15
    v8bf a10 = *(const v8bf*)(Arow1 + k0);
    v8bf a11 = *(const v8bf*)(Arow1 + k0 + 8);
    v16bf af0, af1;
    #pragma unroll
    for (int t = 0; t < 8; ++t) {
      af0[t] = a00[t]; af0[t + 8] = a01[t];
      af1[t] = a10[t]; af1[t + 8] = a11[t];
    }
    #pragma unroll
    for (int nt = 0; nt < 4; ++nt) {
      v16bf bfrag = *(const v16bf*)(Wl[nt] + k0); // K = half*16 + 0..15, contiguous
      acc[nt] = __builtin_amdgcn_wmma_f32_16x16x32_bf16(
          false, af0, false, bfrag, (short)0, acc[nt], false, false);
      acc[4 + nt] = __builtin_amdgcn_wmma_f32_16x16x32_bf16(
          false, af1, false, bfrag, (short)0, acc[4 + nt], false, false);
    }
  }

  #pragma unroll
  for (int mt = 0; mt < 2; ++mt) {
    #pragma unroll
    for (int nt = 0; nt < 4; ++nt) {
      #pragma unroll
      for (int r = 0; r < 8; ++r) {
        int mrow = mt * 16 + r + 8 * half;        // C/D layout: VGPR r -> M = r + 8*(lane/16)
        int atom = order[row0 + mrow];
        if (atom >= 0) {
          float xv = acc[mt * 4 + nt][r];
          float yv = xv / (1.0f + __expf(-xv));   // SiLU
          Out[(size_t)atom * HID + nbase + nt * 16 + nB] = f32_to_bf16(yv);
        }
      }
    }
  }
}

// ---------------------------------------------------------------- layer 3 + reduction
__global__ void layer3_kernel(const unsigned short* __restrict__ h2,
                              const float* __restrict__ W3,
                              const int* __restrict__ numbers, float* __restrict__ eat) {
  int a = blockIdx.x * 256 + threadIdx.x;
  if (a >= N_ATOMS) return;
  const float* w = W3 + numbers[a] * HID;
  const unsigned short* hr = h2 + (size_t)a * HID;
  float sum = 0.0f;
  for (int k = 0; k < HID; ++k) sum += bf16_to_f32(hr[k]) * w[k];
  eat[a] = sum;
}

__global__ void reduce_struct_kernel(const float* __restrict__ eat,
                                     const int* __restrict__ batch,
                                     const int* __restrict__ numbers,
                                     const float* __restrict__ Wc, float* __restrict__ out) {
  __shared__ float sm[256];
  int st = blockIdx.x;
  float s = 0.0f;
  for (int a = threadIdx.x; a < N_ATOMS; a += 256)
    if (batch[a] == st) s += Wc[numbers[a]] + eat[a];   // SCALE == 1.0
  sm[threadIdx.x] = s;
  __syncthreads();
  for (int w = 128; w > 0; w >>= 1) {
    if (threadIdx.x < w) sm[threadIdx.x] += sm[threadIdx.x + w];
    __syncthreads();
  }
  if (threadIdx.x == 0) out[st] = sm[0];
}

// ---------------------------------------------------------------- launch
static inline size_t al256(size_t x) { return (x + 255) & ~(size_t)255; }

extern "C" void kernel_launch(void* const* d_in, const int* in_sizes, int n_in,
                              void* d_out, int out_size, void* d_ws, size_t ws_size,
                              hipStream_t stream) {
  const float* positions    = (const float*)d_in[0];
  const float* cells        = (const float*)d_in[1];
  const int*   numbers      = (const int*)d_in[2];
  const int*   edge_indices = (const int*)d_in[3];
  const int*   edge_offsets = (const int*)d_in[4];
  const int*   batch        = (const int*)d_in[5];
  const float* Wc           = (const float*)d_in[6];
  const float* W1           = (const float*)d_in[7];
  const float* W2           = (const float*)d_in[8];
  const float* W3           = (const float*)d_in[9];
  float* out = (float*)d_out;

  // workspace layout (~235 MB total)
  char* ws = (char*)d_ws;
  size_t o = 0;
  float*          c_buf = (float*)(ws + o);          o = al256(o + (size_t)N_ATOMS * QDIM * NSPH * 4);
  unsigned short* psB   = (unsigned short*)(ws + o); o = al256(o + (size_t)N_ATOMS * PS_DIM * 2);
  unsigned short* h1    = (unsigned short*)(ws + o); o = al256(o + (size_t)N_ATOMS * HID * 2);
  unsigned short* h2    = (unsigned short*)(ws + o); o = al256(o + (size_t)N_ATOMS * HID * 2);
  float*          eat   = (float*)(ws + o);          o = al256(o + (size_t)N_ATOMS * 4);
  unsigned short* w1t   = (unsigned short*)(ws + o); o = al256(o + (size_t)N_SPECIES * PS_DIM * HID * 2);
  unsigned short* w2t   = (unsigned short*)(ws + o); o = al256(o + (size_t)N_SPECIES * HID * HID * 2);
  int*            order = (int*)(ws + o);            o = al256(o + (size_t)ORDER_PAD * 4);
  int*            meta  = (int*)(ws + o);            o = al256(o + 64);
  (void)ws_size; (void)in_sizes; (void)n_in; (void)out_size;

  const int nc = N_ATOMS * QDIM * NSPH;
  zero_f32_kernel<<<(nc + 255) / 256, 256, 0, stream>>>(c_buf, nc);
  init_meta_order_kernel<<<(ORDER_PAD + 255) / 256, 256, 0, stream>>>(meta, order);
  bucket_count_kernel<<<(N_ATOMS + 255) / 256, 256, 0, stream>>>(numbers, meta);
  bucket_offsets_kernel<<<1, 1, 0, stream>>>(meta);
  bucket_fill_kernel<<<(N_ATOMS + 255) / 256, 256, 0, stream>>>(numbers, meta, order);

  edge_kernel<<<(N_EDGES + 255) / 256, 256, 0, stream>>>(
      positions, cells, numbers, edge_indices, edge_offsets, batch, c_buf);

  ps_kernel<<<N_ATOMS, 256, 0, stream>>>(c_buf, psB);

  w_transpose_bf16_kernel<<<(N_SPECIES * PS_DIM * HID + 255) / 256, 256, 0, stream>>>(
      W1, w1t, PS_DIM, HID);
  w_transpose_bf16_kernel<<<(N_SPECIES * HID * HID + 255) / 256, 256, 0, stream>>>(
      W2, w2t, HID, HID);

  gemm_species_silu_kernel<<<GEMM_GRID, 256, 0, stream>>>(psB, w1t, h1, order, meta, PS_DIM);
  gemm_species_silu_kernel<<<GEMM_GRID, 256, 0, stream>>>(h1, w2t, h2, order, meta, HID);

  layer3_kernel<<<(N_ATOMS + 255) / 256, 256, 0, stream>>>(h2, W3, numbers, eat);
  reduce_struct_kernel<<<N_STRUCT, 256, 0, stream>>>(eat, batch, numbers, Wc, out);
}
